// Adaptive_Tiling_Transformer_Block_15848429322486
// MI455X (gfx1250) — compile-verified, matrix operand order checked by reading, not run on hardware
//
#include <hip/hip_runtime.h>

typedef __attribute__((ext_vector_type(16))) _Float16 v16h;
typedef __attribute__((ext_vector_type(8)))  _Float16 v8h;
typedef __attribute__((ext_vector_type(8)))  float    v8f;

#define HW2 192
#define NPIX (HW2*HW2)        // 36864
#define NT 47
#define C64 64
#define HF 128

__device__ __forceinline__ v8f wmma16(v16h a, v16h b, v8f c) {
  return __builtin_amdgcn_wmma_f32_16x16x32_f16(false, a, false, b, (short)0, c, false, false);
}

// K index held by vector element e for the 16-bit A/B fragment layout (ISA 7.12.2)
__device__ __forceinline__ int frag_k(int e, int lane) {
  int v = e >> 1, hb = e & 1;
  int kh = (lane & 16) ? 8 : 0;
  return (v < 4) ? (kh + 2*v + hb) : (16 + kh + 2*(v - 4) + hb);
}

// A fragment (16x32): two contiguous 16B runs per lane -> 2x ds_load_b128.
// Requires ld % 8 == 0 and 16B-aligned base.
__device__ __forceinline__ v16h frag_a(const _Float16* base, int ld, int m0, int k0, int lane) {
  int m = m0 + (lane & 15);
  int kh = (lane & 16) ? 8 : 0;
  const _Float16* p = base + m * ld + k0 + kh;
  v8h lo = *(const v8h*)p;
  v8h hi = *(const v8h*)(p + 16);
  return __builtin_shufflevector(lo, hi, 0,1,2,3,4,5,6,7,8,9,10,11,12,13,14,15);
}

// B fragment from pre-packed weights: 32 contiguous bytes per lane.
__device__ __forceinline__ v16h frag_b_pk(const _Float16* pk, int tile, int lane) {
  return *(const v16h*)(pk + ((size_t)tile * 32 + lane) * 16);
}

// A fragment for one head of Q: A[m][k] = (k<8) ? Q[m0+m][h*8+k] : 0
__device__ __forceinline__ v16h frag_a_head(const _Float16 (*Q)[72], int m0, int h, int lane) {
  int m = m0 + (lane & 15);
  v16h a;
#pragma unroll
  for (int e = 0; e < 16; ++e) { int k = frag_k(e, lane); a[e] = (k < 8) ? Q[m][h*8+k] : (_Float16)0.f; }
  return a;
}
// B fragment for K^T of one head: B[k][n] = (k<8) ? K[n0+n][h*8+k] : 0
__device__ __forceinline__ v16h frag_bT_head(const _Float16 (*Km)[72], int n0, int h, int lane) {
  int n = n0 + (lane & 15);
  v16h b;
#pragma unroll
  for (int e = 0; e < 16; ++e) { int k = frag_k(e, lane); b[e] = (k < 8) ? Km[n][h*8+k] : (_Float16)0.f; }
  return b;
}
// B fragment from V columns of head h (cols n>=8 wrap; outputs discarded)
__device__ __forceinline__ v16h frag_b_vhead(const _Float16 (*V)[72], int k0, int h, int lane) {
  int n = lane & 7;
  v16h b;
#pragma unroll
  for (int e = 0; e < 16; ++e) { int k = frag_k(e, lane); b[e] = V[k0 + k][h*8 + n]; }
  return b;
}

__device__ __forceinline__ void store_tile_h(_Float16 (*D)[72], int m0, int n0, v8f c, int lane, float s) {
  int n = n0 + (lane & 15), mo = (lane & 16) ? 8 : 0;
#pragma unroll
  for (int i = 0; i < 8; ++i) D[m0 + mo + i][n] = (_Float16)(c[i] * s);
}

// ---------------- TDM helpers ----------------
#if __has_builtin(__builtin_amdgcn_tensor_load_to_lds)
#define HAVE_TDM 1
typedef unsigned int u32x4 __attribute__((ext_vector_type(4)));
typedef int i32x8 __attribute__((ext_vector_type(8)));
typedef int i32x4 __attribute__((ext_vector_type(4)));

__device__ __forceinline__ unsigned lds_off(const void* p) {
  return (unsigned)(unsigned long long)p;   // low 32 bits of generic LDS addr = byte offset (ISA 10.2)
}

// 1-D contiguous copy of nelem f16 with LDS row padding (padi: 2^(padi+1) DW interval, padAmt code = DW-1)
__device__ __forceinline__ void tdm_copy_f16(unsigned dst, const void* src, unsigned nelem,
                                             unsigned padi, unsigned padAmt) {
  unsigned long long ga = (unsigned long long)src;
  u32x4 g0; g0[0] = 1u; g0[1] = dst;
  g0[2] = (unsigned)ga;
  g0[3] = (unsigned)((ga >> 32) & 0x01FFFFFFu) | (2u << 30);
  i32x8 g1;
  g1[0] = (int)((1u << 16) | (1u << 20) | (padi << 22) | (padAmt << 25)); // 2B elems, pad
  g1[1] = (int)((nelem & 0xFFFFu) << 16);                 // tensor_dim0 lo16
  g1[2] = (int)(((nelem >> 16) & 0xFFFFu) | (1u << 16));  // dim0 hi | tensor_dim1=1
  g1[3] = (int)((nelem & 0xFFFFu) << 16);                 // tile_dim0
  g1[4] = 1;                                              // tile_dim1=1
  g1[5] = (int)nelem; g1[6] = 0; g1[7] = 0;
  i32x4 z4 = {0,0,0,0};
  i32x8 z8 = {0,0,0,0,0,0,0,0};
  __builtin_amdgcn_tensor_load_to_lds(g0, g1, z4, z4, z8, 0);
}

// 3-D tile 8x8x64 f32 (x,y,ch; strides 1/192/36864), pad 2 DW per 64 DW -> ld 66 f32
__device__ __forceinline__ void tdm_copy_tile3d_f32(unsigned dst, const void* src) {
  unsigned long long ga = (unsigned long long)src;
  u32x4 g0; g0[0] = 1u; g0[1] = dst;
  g0[2] = (unsigned)ga;
  g0[3] = (unsigned)((ga >> 32) & 0x01FFFFFFu) | (2u << 30);
  i32x8 g1;
  g1[0] = (int)((2u << 16) | (1u << 20) | (5u << 22) | (1u << 25)); // 4B elems, pad 2DW/64DW
  g1[1] = (int)(192u << 16);                // tensor_dim0 = 192
  g1[2] = (int)(192u << 16);                // dim0 hi=0 | tensor_dim1 = 192
  g1[3] = (int)(8u << 16);                  // dim1 hi=0 | tile_dim0 = 8
  g1[4] = (int)(8u | (64u << 16));          // tile_dim1 = 8 | tile_dim2 = 64
  g1[5] = 192;                              // tensor_dim0_stride
  g1[6] = (int)((36864u & 0xFFFFu) << 16);  // dim0str hi=0 | dim1_stride lo16
  g1[7] = (int)(36864u >> 16);              // dim1_stride hi
  i32x4 g2 = {64, 0, 0, 0};                 // tensor_dim2 = 64
  i32x4 g3 = {0,0,0,0};
  i32x8 z8 = {0,0,0,0,0,0,0,0};
  __builtin_amdgcn_tensor_load_to_lds(g0, g1, g2, g3, z8, 0);
}
#endif

// ---------------- K0: pack weights into fragment-native f16 tiles ----------------
// pk layout: [wq 8][wk 8][wv 8][wo 8][win 32][wout 16] tiles; tile = 32 lanes x 16 halves
__global__ void k_pack(const float* wq, const float* wk, const float* wv, const float* wo,
                       const float* win, const float* wout, _Float16* pk) {
  int t = blockIdx.x * blockDim.x + threadIdx.x;
  if (t >= 80 * 32) return;
  int tile = t >> 5, lane = t & 31;
  const float* W; int N, loc;
  if (tile < 8)        { W = wq;   N = 64;  loc = tile; }
  else if (tile < 16)  { W = wk;   N = 64;  loc = tile - 8; }
  else if (tile < 24)  { W = wv;   N = 64;  loc = tile - 16; }
  else if (tile < 32)  { W = wo;   N = 64;  loc = tile - 24; }
  else if (tile < 64)  { W = win;  N = 256; loc = tile - 32; }
  else                 { W = wout; N = 64;  loc = tile - 64; }
  int NTl = N / 16;
  int kt = loc / NTl, nt = loc % NTl;
  _Float16* dst = pk + ((size_t)tile * 32 + lane) * 16;
#pragma unroll
  for (int e = 0; e < 16; ++e)
    dst[e] = (_Float16)W[(kt * 32 + frag_k(e, lane)) * N + nt * 16 + (lane & 15)];
}

// ---------------- K1: per-tile attention ----------------
__global__ __launch_bounds__(256) void k_tile_attn(
    const float* __restrict__ x, const float* __restrict__ ln1g, const float* __restrict__ ln1b,
    const _Float16* __restrict__ pk, float* __restrict__ accg) {
  __shared__ __align__(16) float    sTc[64][66];  // residual, [ch][token] (TDM order)
  __shared__ __align__(16) _Float16 sA[64][72];   // LN out / softmax probs
  __shared__ __align__(16) _Float16 sQ[64][72];
  __shared__ __align__(16) _Float16 sK[64][72];
  __shared__ __align__(16) _Float16 sV[64][72];
  __shared__ __align__(16) _Float16 sO[64][72];

  const _Float16* wqP = pk;
  const _Float16* wkP = pk + 8 * 512;
  const _Float16* wvP = pk + 16 * 512;
  const _Float16* woP = pk + 24 * 512;

  int tile = blockIdx.x;
  int batch = tile / (NT * NT);
  int r = tile % (NT * NT);
  int y0 = (r / NT) * 4, x0 = (r % NT) * 4;

  int tid = threadIdx.x;
  int lane = tid & 31, wvid = tid >> 5;

  // 1) tile gather (8x8x64 f32) via Tensor Data Mover
#if defined(HAVE_TDM)
  if (wvid == 0) {
    tdm_copy_tile3d_f32(lds_off(&sTc[0][0]),
                        x + (size_t)batch * C64 * NPIX + y0 * HW2 + x0);
    __builtin_amdgcn_s_wait_tensorcnt(0);
  }
#else
  for (int i = tid; i < 64 * 64; i += 256) {
    int ch = i >> 6, p = i & 63;
    sTc[ch][p] = x[(size_t)(batch * C64 + ch) * NPIX + (y0 + (p >> 3)) * HW2 + x0 + (p & 7)];
  }
#endif
  __syncthreads();

  // 2) LayerNorm per token: 4 threads per row, shfl reductions within 4-lane groups
  {
    int row = tid >> 2, qq = tid & 3;
    float s1 = 0.f, s2 = 0.f;
#pragma unroll
    for (int j = 0; j < 16; ++j) {
      float v = sTc[qq * 16 + j][row];
      s1 += v; s2 += v * v;
    }
    s1 += __shfl_xor(s1, 1, 32); s1 += __shfl_xor(s1, 2, 32);
    s2 += __shfl_xor(s2, 1, 32); s2 += __shfl_xor(s2, 2, 32);
    float mu = s1 * (1.f / 64.f);
    float var = s2 * (1.f / 64.f) - mu * mu;
    float rs = rsqrtf(var + 1e-5f);
#pragma unroll
    for (int j = 0; j < 16; ++j) {
      int c2 = qq * 16 + j;
      sA[row][c2] = (_Float16)((sTc[c2][row] - mu) * rs * ln1g[c2] + ln1b[c2]);
    }
  }
  __syncthreads();

  // 3) Q, K, V projections
  for (int mm = 0; mm < 3; ++mm) {
    const _Float16* W = (mm == 0) ? wqP : (mm == 1) ? wkP : wvP;
    _Float16 (*dst)[72] = (mm == 0) ? sQ : (mm == 1) ? sK : sV;
    float scl = (mm == 0) ? 0.35355339059f : 1.f;   // 1/sqrt(dh) into Q
    for (int t2 = 0; t2 < 2; ++t2) {
      int tt = wvid * 2 + t2, mt = tt >> 2, nt = tt & 3;
      v8f acc = {};
      for (int kt = 0; kt < 2; ++kt) {
        v16h a = frag_a(&sA[0][0], 72, mt * 16, kt * 32, lane);
        v16h b = frag_b_pk(W, kt * 4 + nt, lane);
        acc = wmma16(a, b, acc);
      }
      store_tile_h(dst, mt * 16, nt * 16, acc, lane, scl);
    }
  }
  __syncthreads();

  // 4) heads sequential
  for (int h = 0; h < 8; ++h) {
    for (int t2 = 0; t2 < 2; ++t2) {
      int tt = wvid * 2 + t2, mt = tt >> 2, nt = tt & 3;
      v16h a = frag_a_head(sQ, mt * 16, h, lane);
      v16h b = frag_bT_head(sK, nt * 16, h, lane);
      v8f s = {};
      s = wmma16(a, b, s);
      store_tile_h(sA, mt * 16, nt * 16, s, lane, 1.f);
    }
    __syncthreads();
    {   // softmax: 4 threads/row
      int row = tid >> 2, qq = tid & 3;
      float mx = -1e30f;
#pragma unroll
      for (int j = 0; j < 16; ++j) mx = fmaxf(mx, (float)sA[row][qq * 16 + j]);
      mx = fmaxf(mx, __shfl_xor(mx, 1, 32));
      mx = fmaxf(mx, __shfl_xor(mx, 2, 32));
      float sum = 0.f;
#pragma unroll
      for (int j = 0; j < 16; ++j) {
        int c2 = qq * 16 + j;
        float e = __expf((float)sA[row][c2] - mx);
        sum += e;
        sA[row][c2] = (_Float16)e;
      }
      sum += __shfl_xor(sum, 1, 32); sum += __shfl_xor(sum, 2, 32);
      float inv = 1.f / sum;
#pragma unroll
      for (int j = 0; j < 16; ++j) {
        int c2 = qq * 16 + j;
        sA[row][c2] = (_Float16)((float)sA[row][c2] * inv);
      }
    }
    __syncthreads();
    if (wvid < 4) {   // O_head = P @ V_h (N cols 8..15 discarded)
      int mt = wvid;
      v8f o = {};
      for (int kt = 0; kt < 2; ++kt) {
        v16h a = frag_a(&sA[0][0], 72, mt * 16, kt * 32, lane);
        v16h b = frag_b_vhead(sV, kt * 32, h, lane);
        o = wmma16(a, b, o);
      }
      int n = lane & 15, mo = (lane & 16) ? 8 : 0;
      if (n < 8) {
#pragma unroll
        for (int i = 0; i < 8; ++i) sO[mt * 16 + mo + i][h * 8 + n] = (_Float16)o[i];
      }
    }
    __syncthreads();
  }

  // 5) t_out = O @ wo + t ; overlap-add scatter with f32 atomics
  for (int t2 = 0; t2 < 2; ++t2) {
    int tt = wvid * 2 + t2, mt = tt >> 2, nt = tt & 3;
    int n = lane & 15, mo = (lane & 16) ? 8 : 0;
    v8f c;
#pragma unroll
    for (int i = 0; i < 8; ++i) c[i] = sTc[nt * 16 + n][mt * 16 + mo + i];   // residual as C
    for (int kt = 0; kt < 2; ++kt) {
      v16h a = frag_a(&sO[0][0], 72, mt * 16, kt * 32, lane);
      v16h b = frag_b_pk(woP, kt * 4 + nt, lane);
      c = wmma16(a, b, c);
    }
#pragma unroll
    for (int i = 0; i < 8; ++i) {
      int p = mt * 16 + mo + i, ch = nt * 16 + n;
      atomicAdd(&accg[(size_t)(batch * C64 + ch) * NPIX + (y0 + (p >> 3)) * HW2 + x0 + (p & 7)], c[i]);
    }
  }
}

// ---------------- K2: overlap normalize + LN2 ----------------
__device__ __forceinline__ int coverage(int y) {
  int c = 0, lo = y - 7; if (lo < 0) lo = 0;
  int hi = y; if (hi > 184) hi = 184;
  for (int s = lo; s <= hi; ++s) if (!(s & 3)) ++c;
  return c;
}

__global__ void k_agg_ln2(const float* __restrict__ accg,
                          const float* __restrict__ ln2g, const float* __restrict__ ln2b,
                          float* __restrict__ tokg, _Float16* __restrict__ tokn16) {
  int idx = blockIdx.x * blockDim.x + threadIdx.x;
  if (idx >= 2 * NPIX) return;
  int b = idx / NPIX, pix = idx % NPIX;
  int y = pix / HW2, xx = pix % HW2;
  float inv = 1.f / (float)(coverage(y) * coverage(xx));
  float vals[64];
  float mu = 0.f;
#pragma unroll 8
  for (int ch = 0; ch < 64; ++ch) {
    float v = accg[(size_t)(b * C64 + ch) * NPIX + pix] * inv;
    vals[ch] = v; mu += v;
  }
  mu *= (1.f / 64.f);
  float var = 0.f;
#pragma unroll 8
  for (int ch = 0; ch < 64; ++ch) { float d = vals[ch] - mu; var += d * d; }
  var *= (1.f / 64.f);
  float rs = rsqrtf(var + 1e-5f);
#pragma unroll 8
  for (int ch = 0; ch < 64; ++ch) {
    tokg[(size_t)idx * 64 + ch] = vals[ch];
    tokn16[(size_t)idx * 64 + ch] = (_Float16)((vals[ch] - mu) * rs * ln2g[ch] + ln2b[ch]);
  }
}

// ---------------- K3: u = tokn @ w_in  (M=73728, K=64, N=256) ----------------
__global__ __launch_bounds__(256) void k_gemm_in(const _Float16* __restrict__ tokn16,
                                                 const _Float16* __restrict__ pk,
                                                 _Float16* __restrict__ u16) {
  __shared__ __align__(16) _Float16 sAa[64][72];
  const _Float16* winP = pk + 32 * 512;
  int m0 = blockIdx.x * 64, nb = blockIdx.y;
  int tid = threadIdx.x, lane = tid & 31, wvid = tid >> 5;
#if defined(HAVE_TDM)
  if (wvid == 0) {
    tdm_copy_f16(lds_off(&sAa[0][0]), tokn16 + (size_t)m0 * 64, 4096, 4, 3); // pad 4DW per 32DW -> ld 72
    __builtin_amdgcn_s_wait_tensorcnt(0);
  }
#else
  for (int i = tid; i < 64 * 8; i += 256) {
    int row = i >> 3, c8 = i & 7;
    *(v8h*)&sAa[row][c8 * 8] = *(const v8h*)&tokn16[(size_t)(m0 + row) * 64 + c8 * 8];
  }
#endif
  __syncthreads();
  for (int t2 = 0; t2 < 2; ++t2) {
    int tt = wvid * 2 + t2, mt = tt >> 2, nt = tt & 3;
    v8f acc = {};
    for (int kt = 0; kt < 2; ++kt) {
      v16h a = frag_a(&sAa[0][0], 72, mt * 16, kt * 32, lane);
      v16h b = frag_b_pk(winP, kt * 16 + nb * 4 + nt, lane);
      acc = wmma16(a, b, acc);
    }
    int n = lane & 15, mo = (lane & 16) ? 8 : 0;
#pragma unroll
    for (int i = 0; i < 8; ++i) {
      int m = m0 + mt * 16 + mo + i;
      int b_ = m / NPIX, pix = m % NPIX;
      int ch = nb * 64 + nt * 16 + n;
      u16[(size_t)(b_ * 256 + ch) * NPIX + pix] = (_Float16)acc[i];
    }
  }
}

// ---------------- K4: depthwise 3x3 conv + GELU gate ----------------
__global__ void k_dwconv_gate(const _Float16* __restrict__ u16, const float* __restrict__ wdw,
                              _Float16* __restrict__ z16) {
  int idx = blockIdx.x * blockDim.x + threadIdx.x;
  if (idx >= 2 * HF * NPIX) return;
  int b = idx / (HF * NPIX);
  int r = idx % (HF * NPIX);
  int ch = r / NPIX, pix = r % NPIX;
  int y = pix / HW2, xx = pix % HW2;
  float a1 = 0.f, a2 = 0.f;
#pragma unroll
  for (int ky = 0; ky < 3; ++ky) {
#pragma unroll
    for (int kx = 0; kx < 3; ++kx) {
      int yy = y + ky - 1, x2 = xx + kx - 1;
      if (yy >= 0 && yy < HW2 && x2 >= 0 && x2 < HW2) {
        int off = yy * HW2 + x2;
        a1 += wdw[ch * 9 + ky * 3 + kx]        * (float)u16[(size_t)(b * 256 + ch) * NPIX + off];
        a2 += wdw[(ch + HF) * 9 + ky * 3 + kx] * (float)u16[(size_t)(b * 256 + ch + HF) * NPIX + off];
      }
    }
  }
  float g = 0.5f * a1 * (1.f + erff(a1 * 0.70710678118f));
  z16[(size_t)(b * NPIX + pix) * HF + ch] = (_Float16)(g * a2);
}

// ---------------- K5: out = z @ w_out + tok  (M=73728, K=128, N=64) ----------------
__global__ __launch_bounds__(256) void k_gemm_out(const _Float16* __restrict__ z16,
                                                  const _Float16* __restrict__ pk,
                                                  const float* __restrict__ tokg,
                                                  float* __restrict__ out) {
  __shared__ __align__(16) _Float16 sAa[64][136];
  const _Float16* woutP = pk + 64 * 512;
  int m0 = blockIdx.x * 64;
  int tid = threadIdx.x, lane = tid & 31, wvid = tid >> 5;
#if defined(HAVE_TDM)
  if (wvid == 0) {
    tdm_copy_f16(lds_off(&sAa[0][0]), z16 + (size_t)m0 * HF, 8192, 5, 3); // pad 4DW per 64DW -> ld 136
    __builtin_amdgcn_s_wait_tensorcnt(0);
  }
#else
  for (int i = tid; i < 64 * 16; i += 256) {
    int row = i >> 4, c8 = i & 15;
    *(v8h*)&sAa[row][c8 * 8] = *(const v8h*)&z16[(size_t)(m0 + row) * HF + c8 * 8];
  }
#endif
  __syncthreads();
  for (int t2 = 0; t2 < 2; ++t2) {
    int tt = wvid * 2 + t2, mt = tt >> 2, nt = tt & 3;
    int n = lane & 15, mo = (lane & 16) ? 8 : 0;
    v8f c;
#pragma unroll
    for (int i = 0; i < 8; ++i)
      c[i] = tokg[(size_t)(m0 + mt * 16 + mo + i) * 64 + nt * 16 + n];
    for (int kt = 0; kt < 4; ++kt) {
      v16h a = frag_a(&sAa[0][0], 136, mt * 16, kt * 32, lane);
      v16h b = frag_b_pk(woutP, kt * 4 + nt, lane);
      c = wmma16(a, b, c);
    }
#pragma unroll
    for (int i = 0; i < 8; ++i) {
      int m = m0 + mt * 16 + mo + i;
      int b_ = m / NPIX, pix = m % NPIX;
      int ch = nt * 16 + n;
      out[(size_t)(b_ * C64 + ch) * NPIX + pix] = c[i];
    }
  }
}

extern "C" void kernel_launch(void* const* d_in, const int* in_sizes, int n_in,
                              void* d_out, int out_size, void* d_ws, size_t ws_size,
                              hipStream_t stream) {
  const float* x    = (const float*)d_in[0];
  const float* ln1g = (const float*)d_in[1];
  const float* ln1b = (const float*)d_in[2];
  const float* wq   = (const float*)d_in[3];
  const float* wk   = (const float*)d_in[4];
  const float* wvp  = (const float*)d_in[5];
  const float* wo   = (const float*)d_in[6];
  const float* ln2g = (const float*)d_in[7];
  const float* ln2b = (const float*)d_in[8];
  const float* win  = (const float*)d_in[9];
  const float* wdw  = (const float*)d_in[10];
  const float* wout = (const float*)d_in[11];
  float* out = (float*)d_out;

  char* ws = (char*)d_ws;
  const size_t ACC_BYTES = (size_t)2 * C64 * NPIX * sizeof(float);   // 18.87 MB
  float*    acc    = (float*)ws;
  float*    tokg   = (float*)(ws + ACC_BYTES);
  _Float16* tokn16 = (_Float16*)(ws + 2 * ACC_BYTES);
  _Float16* u16    = (_Float16*)(ws + 2 * ACC_BYTES + ACC_BYTES / 2);
  _Float16* z16    = (_Float16*)ws;                                  // aliases acc (dead by then)
  _Float16* pk     = (_Float16*)(ws + 2 * ACC_BYTES + ACC_BYTES / 2 +
                                 (size_t)2 * 256 * NPIX * sizeof(_Float16));  // 80 KB packed weights

  k_pack<<<10, 256, 0, stream>>>(wq, wk, wvp, wo, win, wout, pk);
  (void)hipMemsetAsync(acc, 0, ACC_BYTES, stream);

  k_tile_attn<<<2 * NT * NT, 256, 0, stream>>>(x, ln1g, ln1b, pk, acc);

  k_agg_ln2<<<(2 * NPIX + 255) / 256, 256, 0, stream>>>(acc, ln2g, ln2b, tokg, tokn16);

  dim3 g3(2 * NPIX / 64, 4);
  k_gemm_in<<<g3, 256, 0, stream>>>(tokn16, pk, u16);

  k_dwconv_gate<<<(2 * HF * NPIX + 255) / 256, 256, 0, stream>>>(u16, wdw, z16);

  k_gemm_out<<<2 * NPIX / 64, 256, 0, stream>>>(z16, pk, tokg, out);
}
